// MultiHeadAttention_80711025427120
// MI455X (gfx1250) — matrix operand order repaired and checked by reading, not compiled, not run on hardware
//
#include <hip/hip_runtime.h>

// ---------------------------------------------------------------------------
// MHA forward for MI455X (gfx1250), bf16 WMMA path with register blocking.
//   B=2, S=2048, D=1024, H=16, HD=64
// ---------------------------------------------------------------------------

#define BB 2
#define SS 2048
#define DD 1024
#define HH 16
#define HD 64

typedef __bf16 bf16_t;
typedef __attribute__((ext_vector_type(16))) __bf16 v16bf;
typedef __attribute__((ext_vector_type(8)))  __bf16 v8bf;
typedef __attribute__((ext_vector_type(4)))  __bf16 v4bf;
typedef __attribute__((ext_vector_type(8)))  float  v8f;

__device__ __forceinline__ v16bf concat8(v8bf lo, v8bf hi) {
    return __builtin_shufflevector(lo, hi, 0,1,2,3,4,5,6,7,8,9,10,11,12,13,14,15);
}

__device__ __forceinline__ v8f wmma_bf16(v16bf a, v16bf b, v8f c) {
    // D = A(16x32 bf16) * B(32x16 bf16) + C(16x16 f32)
    return __builtin_amdgcn_wmma_f32_16x16x32_bf16(
        /*neg_a=*/false, a, /*neg_b=*/false, b,
        /*c_mod=*/(short)0, c, /*reuse_a=*/false, /*reuse_b=*/false);
}

// Reductions across the 16-lane half of a wave32 (rows of a C fragment live
// entirely within one 16-lane half; xor masks 1,2,4,8 stay inside it).
__device__ __forceinline__ float rowmax16(float v) {
    v = fmaxf(v, __shfl_xor(v, 1, 32));
    v = fmaxf(v, __shfl_xor(v, 2, 32));
    v = fmaxf(v, __shfl_xor(v, 4, 32));
    v = fmaxf(v, __shfl_xor(v, 8, 32));
    return v;
}
__device__ __forceinline__ float rowsum16(float v) {
    v += __shfl_xor(v, 1, 32);
    v += __shfl_xor(v, 2, 32);
    v += __shfl_xor(v, 4, 32);
    v += __shfl_xor(v, 8, 32);
    return v;
}

// ---------------------------------------------------------------------------
// f32 -> bf16 conversion, 4 elements / thread
// ---------------------------------------------------------------------------
__global__ __launch_bounds__(256)
void f32_to_bf16_x4(const float* __restrict__ in, bf16_t* __restrict__ out, int n4) {
    int i = blockIdx.x * 256 + threadIdx.x;
    if (i < n4) {
        float4 f = reinterpret_cast<const float4*>(in)[i];
        v4bf o;
        o[0] = (bf16_t)f.x; o[1] = (bf16_t)f.y;
        o[2] = (bf16_t)f.z; o[3] = (bf16_t)f.w;
        reinterpret_cast<v4bf*>(out)[i] = o;
    }
}

// ---------------------------------------------------------------------------
// GEMM: C[M,N] = A[M,K](bf16) * W[N,K]^T (bf16) + bias[N]
// 32x64 output tile per wave32: 2 A-fragments x 4 B-fragments -> 8 WMMAs
// per K-step of 32.  16 consecutive waves share the same A rows.
//   mode 0: store f32 row-major [M,N]                     (output projection)
//   mode 1: store bf16 head-major [B,H,S,HD]              (Q, K)
//   mode 2: store bf16 head-transposed [B,H,HD,S]         (V^T)
// ---------------------------------------------------------------------------
__global__ __launch_bounds__(256)
void gemm_bf16_wmma(const bf16_t* __restrict__ A, const bf16_t* __restrict__ W,
                    const float* __restrict__ bias, void* __restrict__ out,
                    int M, int N, int K, int mode)
{
    const int lane = threadIdx.x & 31;
    const int wid  = blockIdx.x * 8 + (threadIdx.x >> 5);
    const int tilesN = N >> 6;                 // 64-wide N tiles
    const int tm = wid / tilesN;               // 32-row M tile
    const int tn = wid - tm * tilesN;
    const int half = lane >> 4;
    const int l16  = lane & 15;
    const int h8   = half * 8;
    const int h16  = half * 16;

    // A fragments: lane holds rows tm*32+l16 and tm*32+16+l16
    const bf16_t* arow0 = A + (size_t)(tm * 32 + l16) * K;
    const bf16_t* arow1 = arow0 + (size_t)16 * K;
    // B fragments: column (tn*64 + j*16 + l16) of B == that row of W
    const bf16_t* brow0 = W + (size_t)(tn * 64 + l16) * K;

    v8f acc[2][4] = {};
    for (int kk = 0; kk < K; kk += 32) {
        const v16bf a0 = *reinterpret_cast<const v16bf*>(arow0 + kk + h16);
        const v16bf a1 = *reinterpret_cast<const v16bf*>(arow1 + kk + h16);
#pragma unroll
        for (int j = 0; j < 4; ++j) {
            const v16bf b = *reinterpret_cast<const v16bf*>(
                brow0 + (size_t)(j * 16) * K + kk + h16);
            acc[0][j] = wmma_bf16(a0, b, acc[0][j]);
            acc[1][j] = wmma_bf16(a1, b, acc[1][j]);
        }
    }

#pragma unroll
    for (int j = 0; j < 4; ++j) {
        const int   col = tn * 64 + j * 16 + l16;
        const float bv  = bias[col];
        const int   h   = col >> 6, hd = col & (HD - 1);
#pragma unroll
        for (int i = 0; i < 2; ++i) {
#pragma unroll
            for (int v = 0; v < 8; ++v) {
                const int   row = tm * 32 + i * 16 + v + 8 * half;
                const float val = acc[i][j][v] + bv;
                if (mode == 0) {
                    reinterpret_cast<float*>(out)[(size_t)row * N + col] = val;
                } else {
                    const int b = row >> 11, s = row & (SS - 1);
                    if (mode == 1) {
                        reinterpret_cast<bf16_t*>(out)
                            [(((size_t)(b * HH + h) * SS + s) * HD) + hd] = (bf16_t)val;
                    } else {
                        reinterpret_cast<bf16_t*>(out)
                            [(((size_t)(b * HH + h) * HD + hd) * SS) + s] = (bf16_t)val;
                    }
                }
            }
        }
    }
}

// ---------------------------------------------------------------------------
// Flash attention, 32 Q rows (two 16-row tiles) per wave32.  Per 32-key step:
//   8 WMMAs for S = Q*K^T sharing 4 K-fragments
//   online softmax (shfl reductions within 16-lane halves)
//   P staged through per-wave LDS to re-layout C-frag -> A-frag
//   8 WMMAs for O += P*V sharing 4 V-fragments (V stored transposed)
// ---------------------------------------------------------------------------
__global__ __launch_bounds__(256)
void flash_attn_bf16(const bf16_t* __restrict__ q, const bf16_t* __restrict__ k,
                     const bf16_t* __restrict__ vT, bf16_t* __restrict__ o_out)
{
    __shared__ bf16_t lds_p[8][2][16][32];   // per-wave P staging (16 KB)

    const int lane = threadIdx.x & 31;
    const int warp = threadIdx.x >> 5;
    const int wg   = blockIdx.x * 8 + warp;
    const int qt   = wg & (SS / 32 - 1);          // 64 q-blocks (32 rows) per head
    const int h    = (wg >> 6) & (HH - 1);
    const int b    = wg >> 10;
    const int half = lane >> 4;
    const int l16  = lane & 15;
    const int h8   = half * 8;
    const int h16  = half * 16;

    const bf16_t* qbase = q  + (((size_t)(b * HH + h)) * SS + qt * 32) * HD;
    const bf16_t* kbase = k  + ((size_t)(b * HH + h)) * SS * HD;
    const bf16_t* vbase = vT + ((size_t)(b * HH + h)) * HD * SS;

    // Q A-fragments: aQ[qtile][k-chunk], held in registers the whole loop
    v16bf aQ[2][2];
#pragma unroll
    for (int i = 0; i < 2; ++i) {
        const bf16_t* qrow = qbase + (size_t)(i * 16 + l16) * HD;
        aQ[i][0] = *reinterpret_cast<const v16bf*>(qrow + h16);
        aQ[i][1] = *reinterpret_cast<const v16bf*>(qrow + 32 + h16);
    }

    float rm[2][8], rl[2][8];
#pragma unroll
    for (int i = 0; i < 2; ++i)
#pragma unroll
        for (int v = 0; v < 8; ++v) { rm[i][v] = -3.0e38f; rl[i][v] = 0.0f; }
    v8f o[2][4] = {};

    for (int j = 0; j < SS; j += 32) {
        // ---- K fragments: bK[key-subtile jj][d-chunk], shared by both q tiles
        v16bf bK[2][2];
#pragma unroll
        for (int jj = 0; jj < 2; ++jj) {
            const bf16_t* kr = kbase + (size_t)(j + jj * 16 + l16) * HD;
            bK[jj][0] = *reinterpret_cast<const v16bf*>(kr + h16);
            bK[jj][1] = *reinterpret_cast<const v16bf*>(kr + 32 + h16);
        }
        // ---- V fragments: bV[d0/16], shared by both q tiles
        v16bf bV[4];
        {
            const bf16_t* vr = vbase + (size_t)l16 * SS + j + h16;
#pragma unroll
            for (int t = 0; t < 4; ++t)
                bV[t] = *reinterpret_cast<const v16bf*>(vr + (size_t)(t * 16) * SS);
        }

#pragma unroll
        for (int i = 0; i < 2; ++i) {
            // ---- scores for q tile i: c0 = keys j..j+15, c1 = j+16..j+31
            v8f c0 = {}, c1 = {};
            c0 = wmma_bf16(aQ[i][0], bK[0][0], c0);
            c0 = wmma_bf16(aQ[i][1], bK[0][1], c0);
            c1 = wmma_bf16(aQ[i][0], bK[1][0], c1);
            c1 = wmma_bf16(aQ[i][1], bK[1][1], c1);
            // ---- online softmax over the 32 new columns
#pragma unroll
            for (int v = 0; v < 8; ++v) {
                const float s0 = c0[v] * 0.125f;     // 1/sqrt(64)
                const float s1 = c1[v] * 0.125f;
                const float tmax  = rowmax16(fmaxf(s0, s1));
                const float mnew  = fmaxf(rm[i][v], tmax);
                const float scale = __expf(rm[i][v] - mnew);
                rm[i][v] = mnew;
                const float p0 = __expf(s0 - mnew);
                const float p1 = __expf(s1 - mnew);
                rl[i][v] = rl[i][v] * scale + rowsum16(p0 + p1);
                o[i][0][v] *= scale; o[i][1][v] *= scale;
                o[i][2][v] *= scale; o[i][3][v] *= scale;
                const int row = v + 8 * half;
                lds_p[warp][i][row][l16]      = (bf16_t)p0;
                lds_p[warp][i][row][16 + l16] = (bf16_t)p1;
            }
            // re-read P in A-fragment layout (wave-private LDS, DS in-order)
            const v16bf aP = *reinterpret_cast<const v16bf*>(&lds_p[warp][i][l16][h16]);
            // ---- O += P * V
            o[i][0] = wmma_bf16(aP, bV[0], o[i][0]);
            o[i][1] = wmma_bf16(aP, bV[1], o[i][1]);
            o[i][2] = wmma_bf16(aP, bV[2], o[i][2]);
            o[i][3] = wmma_bf16(aP, bV[3], o[i][3]);
        }
    }

    // ---- finalize: O /= rowsum; store bf16 at [b, s, h*64 + d] ----
#pragma unroll
    for (int i = 0; i < 2; ++i) {
#pragma unroll
        for (int v = 0; v < 8; ++v) {
            const float inv  = 1.0f / rl[i][v];
            const int   srow = qt * 32 + i * 16 + v + 8 * half;
            bf16_t* dst = o_out + ((size_t)b * SS + srow) * DD + h * HD + l16;
            dst[0]  = (bf16_t)(o[i][0][v] * inv);
            dst[16] = (bf16_t)(o[i][1][v] * inv);
            dst[32] = (bf16_t)(o[i][2][v] * inv);
            dst[48] = (bf16_t)(o[i][3][v] * inv);
        }
    }
}

// ---------------------------------------------------------------------------
// Host launch
// ---------------------------------------------------------------------------
extern "C" void kernel_launch(void* const* d_in, const int* in_sizes, int n_in,
                              void* d_out, int out_size, void* d_ws, size_t ws_size,
                              hipStream_t stream) {
    const float* x  = (const float*)d_in[0];
    const float* Wq = (const float*)d_in[1];
    const float* bq = (const float*)d_in[2];
    const float* Wk = (const float*)d_in[3];
    const float* bk = (const float*)d_in[4];
    const float* Wv = (const float*)d_in[5];
    const float* bv = (const float*)d_in[6];
    const float* Wo = (const float*)d_in[7];
    const float* bo = (const float*)d_in[8];

    char* ws = (char*)d_ws;
    bf16_t* x_bf   = (bf16_t*)(ws + (0ull  << 20));   //  8 MiB  [B*S, D]
    bf16_t* wq_bf  = (bf16_t*)(ws + (8ull  << 20));   //  2 MiB  [D, D]
    bf16_t* wk_bf  = (bf16_t*)(ws + (10ull << 20));
    bf16_t* wv_bf  = (bf16_t*)(ws + (12ull << 20));
    bf16_t* wo_bf  = (bf16_t*)(ws + (14ull << 20));
    bf16_t* q_bf   = (bf16_t*)(ws + (16ull << 20));   //  8 MiB  [B,H,S,HD]
    bf16_t* k_bf   = (bf16_t*)(ws + (24ull << 20));   //  8 MiB  [B,H,S,HD]
    bf16_t* vT_bf  = (bf16_t*)(ws + (32ull << 20));   //  8 MiB  [B,H,HD,S]
    bf16_t* at_bf  = (bf16_t*)(ws + (40ull << 20));   //  8 MiB  [B*S, D]

    const int M = BB * SS;            // 4096
    const int N = DD, K = DD;         // 1024

    // 1) convert to bf16
    {
        const int nX4 = (M * DD) / 4;         // 1048576
        f32_to_bf16_x4<<<nX4 / 256, 256, 0, stream>>>(x, x_bf, nX4);
        const int nW4 = (DD * DD) / 4;        // 262144
        f32_to_bf16_x4<<<nW4 / 256, 256, 0, stream>>>(Wq, wq_bf, nW4);
        f32_to_bf16_x4<<<nW4 / 256, 256, 0, stream>>>(Wk, wk_bf, nW4);
        f32_to_bf16_x4<<<nW4 / 256, 256, 0, stream>>>(Wv, wv_bf, nW4);
        f32_to_bf16_x4<<<nW4 / 256, 256, 0, stream>>>(Wo, wo_bf, nW4);
    }

    // 2) projections: (4096/32)*(1024/64) = 2048 waves, 8 waves / block
    const int gemm_blocks = (M / 32) * (N / 64) / 8;   // 256
    gemm_bf16_wmma<<<gemm_blocks, 256, 0, stream>>>(x_bf, wq_bf, bq, q_bf,  M, N, K, 1);
    gemm_bf16_wmma<<<gemm_blocks, 256, 0, stream>>>(x_bf, wk_bf, bk, k_bf,  M, N, K, 1);
    gemm_bf16_wmma<<<gemm_blocks, 256, 0, stream>>>(x_bf, wv_bf, bv, vT_bf, M, N, K, 2);

    // 3) flash attention: B*H*(S/32) = 2048 waves -> 256 blocks
    flash_attn_bf16<<<(BB * HH * (SS / 32)) / 8, 256, 0, stream>>>(q_bf, k_bf, vT_bf, at_bf);

    // 4) output projection (f32 out + bias)
    gemm_bf16_wmma<<<gemm_blocks, 256, 0, stream>>>(at_bf, wo_bf, bo, d_out, M, N, K, 0);
}
